// GaussianPolicyLSTM_6365141533306
// MI455X (gfx1250) — matrix operand order repaired and checked by reading, not compiled, not run on hardware
//
#include <hip/hip_runtime.h>
#include <hip/hip_bf16.h>
#include <cstdint>
#include <cstddef>

// ---------------------------------------------------------------------------
// Problem constants (from reference): T=16384, B=1, D=512, H=1024, O=256
// ---------------------------------------------------------------------------
#define TT 16384
#define DD 512
#define HH 1024
#define OO 256
#define H4 (4 * HH)   // 4096

// ---------------------------------------------------------------------------
// CDNA5 WMMA / TDM types
// ---------------------------------------------------------------------------
typedef __bf16        v16bf __attribute__((ext_vector_type(16)));
typedef float         v8f   __attribute__((ext_vector_type(8)));
typedef unsigned int  u32x4 __attribute__((ext_vector_type(4)));
typedef int           i32x4 __attribute__((ext_vector_type(4)));
typedef int           i32x8 __attribute__((ext_vector_type(8)));

union Frag {
    struct { uint4 lo, hi; } u;   // two 16B chunks per lane
    v16bf v;                      // 16 bf16 = 32B per lane (8 VGPRs)
};

__device__ __forceinline__ __bf16 f2bf(float x) {
    __hip_bfloat16 t = __float2bfloat16(x);
    __bf16 r;
    __builtin_memcpy(&r, &t, 2);
    return r;
}

// ---------------------------------------------------------------------------
// TDM: DMA one contiguous block (rows are contiguous -> 2D tile with 1 line)
// from global memory into LDS. units8 = block size in 8-byte elements.
// D# packing per CDNA5 ISA ch.8 (group0: count/lds/global/type,
// group1: data_size=8B, tensor_dim0 = tile_dim0 = units8, tile_dim1 = 1).
// ---------------------------------------------------------------------------
#if __has_builtin(__builtin_amdgcn_tensor_load_to_lds)
#define HAVE_TDM 1
__device__ __forceinline__ void tdm_load_block(unsigned lds_addr, const void* gptr,
                                               unsigned units8) {
    unsigned long long ga = (unsigned long long)gptr;
    u32x4 g0;
    g0[0] = 1u;                                            // count=1 (valid), user mode
    g0[1] = lds_addr;                                      // LDS byte address
    g0[2] = (unsigned)(ga & 0xffffffffu);                  // global_addr[31:0]
    g0[3] = (unsigned)((ga >> 32) & 0x01ffffffu)           // global_addr[56:32]
          | (2u << 30);                                    // type = 2 ("image")
    i32x8 g1;
    g1[0] = (int)(3u << 16);                               // data_size = 3 (8 bytes)
    g1[1] = (int)((units8 & 0xffffu) << 16);               // tensor_dim0[15:0]
    g1[2] = (int)((units8 >> 16) | (1u << 16));            // tensor_dim0[31:16] | tensor_dim1=1
    g1[3] = (int)((units8 & 0xffffu) << 16);               // tile_dim0 = units8
    g1[4] = 1;                                             // tile_dim1 = 1, tile_dim2 = 0
    g1[5] = (int)units8;                                   // tensor_dim0_stride[31:0]
    g1[6] = 0;
    g1[7] = 0;
    i32x4 z4 = {0, 0, 0, 0};
#if __clang_major__ >= 23
    i32x8 z8 = {0, 0, 0, 0, 0, 0, 0, 0};
    __builtin_amdgcn_tensor_load_to_lds(g0, g1, z4, z4, z8, 0);
#else
    __builtin_amdgcn_tensor_load_to_lds(g0, g1, z4, z4, 0);
#endif
}
#else
#define HAVE_TDM 0
#endif

// ---------------------------------------------------------------------------
// fp32 -> bf16 conversion (grid-stride)
// ---------------------------------------------------------------------------
__global__ void k_cvt_bf16(const float* __restrict__ s, __bf16* __restrict__ d, long n) {
    long i = (long)blockIdx.x * blockDim.x + threadIdx.x;
    long stride = (long)gridDim.x * blockDim.x;
    for (; i < n; i += stride) d[i] = f2bf(s[i]);
}

// b = x + y  (precompute bih + bhh)
__global__ void k_addbias(const float* __restrict__ x, const float* __restrict__ y,
                          float* __restrict__ b, int n) {
    int i = blockIdx.x * blockDim.x + threadIdx.x;
    if (i < n) b[i] = x[i] + y[i];
}

__global__ void k_zero(unsigned char* __restrict__ p, long n) {
    long i = (long)blockIdx.x * blockDim.x + threadIdx.x;
    long stride = (long)gridDim.x * blockDim.x;
    for (; i < n; i += stride) p[i] = 0;
}

// ---------------------------------------------------------------------------
// bf16 GEMM:  C[M,N] = A[M,K] * W[N,K]^T (+ bias[n])     (all row-major)
// One 16x16 C-tile per wave, K-loop in steps of 32 via v_wmma_f32_16x16x32_bf16.
// A-frag (ISA 7.12.2): lane l holds row m0+(l&15),
//   K = k0 + (l>=16 ? 8 : 0) + {0..7} and + {16..23}  -> two contiguous 16B loads.
// B-frag: lane l holds column n0+(l&15),
//   K = k0 + (l>=16 ? 8 : 0) + {0..7} and + {16..23}  -> two contiguous 16B loads
//   of row (n0+(l&15)) of W[N,K]  (no transpose needed).
// mode: 0 = bias add; 2 = bias add then exp(0.5*x)  (sigma head)
// ---------------------------------------------------------------------------
__global__ void k_gemm_bf16(const __bf16* __restrict__ A, const __bf16* __restrict__ W,
                            const float* __restrict__ bias, float* __restrict__ C,
                            int M, int N, int K, int mode) {
    const int lane = threadIdx.x & 31;
    const int wave = threadIdx.x >> 5;
    const int ntiles = N >> 4;
    const int tile = blockIdx.x * (blockDim.x >> 5) + wave;
    if (tile >= (M >> 4) * ntiles) return;
    const int m0 = (tile / ntiles) << 4;
    const int n0 = (tile % ntiles) << 4;

    const __bf16* arow = A + (size_t)(m0 + (lane & 15)) * K + ((lane >> 4) << 3);
    const __bf16* wrow = W + (size_t)(n0 + (lane & 15)) * K + ((lane >> 4) << 3);

    v8f acc = {0.f, 0.f, 0.f, 0.f, 0.f, 0.f, 0.f, 0.f};
    for (int k0 = 0; k0 < K; k0 += 32) {
        Frag a, b;
        a.u.lo = *(const uint4*)(arow + k0);
        a.u.hi = *(const uint4*)(arow + k0 + 16);
        b.u.lo = *(const uint4*)(wrow + k0);
        b.u.hi = *(const uint4*)(wrow + k0 + 16);
        __builtin_prefetch(arow + k0 + 256, 0, 1);   // global_prefetch_b8
        acc = __builtin_amdgcn_wmma_f32_16x16x32_bf16(
            false, a.v, false, b.v, (short)0, acc, false, false);
    }

    const int n  = n0 + (lane & 15);
    const int mb = m0 + ((lane >> 4) << 3);
    const float bs = bias ? bias[n] : 0.f;
#pragma unroll
    for (int g = 0; g < 8; ++g) {
        float v = acc[g] + bs;
        if (mode == 2) v = __expf(0.5f * v);
        C[(size_t)(mb + g) * N + n] = v;
    }
}

// ---------------------------------------------------------------------------
// Persistent LSTM scan for one layer.
//   grid = H/16 = 64 blocks, 256 threads (8 waves). Block b owns hidden units
//   [j0, j0+16) -> 64 gate rows of Whh. The 128KB bf16 Whh slice is DMA'd into
//   LDS ONCE by the Tensor Data Mover (4 descriptors, one per contiguous
//   16-row gate block), then all 16384 steps feed WMMA A-fragments from LDS.
//   Wave w: gate tile (w&3), K-half (w>>2). B-frag = h_prev broadcast into all
//   16 columns (mat-vec on the matrix core). K-halves reduced through LDS;
//   threads 0..15 apply gate nonlinearities, update c, publish h (bf16), then
//   all 64 blocks cross a global atomic barrier before the next timestep.
//   Dynamic LDS: 64 rows * 1024 bf16 = 131072 bytes.
// ---------------------------------------------------------------------------
#define SCAN_LDS_BYTES (64 * HH * 2)

__global__ void k_lstm_scan(const float* __restrict__ gx,     // [T, 4H] (bih+bhh folded in)
                            const __bf16* __restrict__ Whh,   // [4H, H] bf16 row-major
                            __bf16* __restrict__ hseq,        // [T, H] bf16 out
                            __bf16* __restrict__ hbuf,        // [H] bf16 live h (pre-zeroed)
                            float* __restrict__ outH,         // [H] h_T
                            float* __restrict__ outC,         // [H] c_T
                            unsigned* __restrict__ syncc,     // pre-zeroed barrier counter
                            int Tn) {
    extern __shared__ __align__(16) char wtile[];     // [64][HH] bf16 Whh slice
    __shared__ __align__(16) unsigned short hsm[HH];  // staged h_prev (bf16 bits)
    __shared__ float psum[4][16];                     // K-half-1 partials
    __shared__ float gbuf[4][16];                     // final gate pre-activations

    const int lane  = threadIdx.x & 31;
    const int wave  = threadIdx.x >> 5;
    const int mtile = wave & 3;    // gate: 0=i 1=f 2=g 3=o
    const int khalf = wave >> 2;   // K half: 0 or 1 (512 each)
    const int j0    = blockIdx.x << 4;
    const int u     = threadIdx.x; // unit index when < 16

    // ---- stage this WG's Whh slice into LDS once (TDM async DMA) ----------
#if HAVE_TDM
    if (wave == 0) {
        const unsigned ldsBase = __builtin_amdgcn_groupstaticsize(); // dyn LDS start
        #pragma unroll
        for (int g = 0; g < 4; ++g) {
            // gate block g: 16 contiguous rows of Whh = 16*HH bf16 = 32KB
            tdm_load_block(ldsBase + (unsigned)g * 16 * HH * 2,
                           Whh + (size_t)(g * HH + j0) * HH,
                           (16 * HH * 2) / 8);
        }
        __builtin_amdgcn_s_wait_tensorcnt(0);          // s_wait_tensorcnt 0
    }
#else
    for (int i = threadIdx.x * 8; i < 64 * HH; i += blockDim.x * 8) {
        int r = i / HH, kx = i % HH;
        *(uint4*)(wtile + (size_t)i * 2) =
            *(const uint4*)(Whh + (size_t)((r >> 4) * HH + j0 + (r & 15)) * HH + kx);
    }
#endif
    __syncthreads();

    float c = 0.f, hlast = 0.f, clast = 0.f;

    // A-frag base inside the LDS tile: row = mtile*16 + (lane&15)
    const __bf16* arowBase = (const __bf16*)wtile
        + (size_t)(mtile * 16 + (lane & 15)) * HH
        + khalf * (HH >> 1) + ((lane >> 4) << 3);

    for (int t = 0; t < Tn; ++t) {
        // stage h_prev (written by other WGs last step) into LDS
        for (int i = threadIdx.x; i < HH; i += blockDim.x)
            hsm[i] = ((const volatile unsigned short*)hbuf)[i];
        __syncthreads();

        v8f acc = {0.f, 0.f, 0.f, 0.f, 0.f, 0.f, 0.f, 0.f};
#pragma unroll 4
        for (int kk = 0; kk < (HH >> 1); kk += 32) {
            Frag a, b;
            a.u.lo = *(const uint4*)(arowBase + kk);          // ds_load_b128
            a.u.hi = *(const uint4*)(arowBase + kk + 16);
            const __bf16* hp = (const __bf16*)hsm
                             + khalf * (HH >> 1) + kk + ((lane >> 4) << 3);
            b.u.lo = *(const uint4*)hp;                       // ds_load_b128
            b.u.hi = *(const uint4*)(hp + 16);
            acc = __builtin_amdgcn_wmma_f32_16x16x32_bf16(
                false, a.v, false, b.v, (short)0, acc, false, false);
        }

        // column-0 extraction (B broadcast -> every column holds the result).
        // lane 0 holds M = 0..7 in acc[0..7], lane 16 holds M = 8..15.
        if ((lane & 15) == 0 && khalf == 1) {
            const int h8 = (lane >> 4) << 3;
#pragma unroll
            for (int g = 0; g < 8; ++g) psum[mtile][h8 + g] = acc[g];
        }
        __syncthreads();
        if ((lane & 15) == 0 && khalf == 0) {
            const int h8 = (lane >> 4) << 3;
#pragma unroll
            for (int g = 0; g < 8; ++g) gbuf[mtile][h8 + g] = acc[g] + psum[mtile][h8 + g];
        }
        __syncthreads();

        if (u < 16) {
            const float* gxr = gx + (size_t)t * H4 + j0 + u;
            float ig = gbuf[0][u] + gxr[0];
            float fg = gbuf[1][u] + gxr[HH];
            float gg = gbuf[2][u] + gxr[2 * HH];
            float og = gbuf[3][u] + gxr[3 * HH];
            ig = 1.f / (1.f + __expf(-ig));
            fg = 1.f / (1.f + __expf(-fg));
            og = 1.f / (1.f + __expf(-og));
            gg = tanhf(gg);
            c = fg * c + ig * gg;
            float h = og * tanhf(c);
            hlast = h; clast = c;
            __bf16 hb = f2bf(h);
            hseq[(size_t)t * HH + j0 + u] = hb;
            hbuf[j0 + u] = hb;
        }
        __syncthreads();

        // grid-wide barrier (monotonic counter, 64 co-resident blocks)
        if (threadIdx.x == 0) {
            __threadfence();
            atomicAdd(syncc, 1u);
            const unsigned target = (unsigned)(t + 1) * gridDim.x;
            while (atomicAdd(syncc, 0u) < target) { }
            __threadfence();
        }
        __syncthreads();
    }

    if (u < 16) {
        outH[j0 + u] = hlast;
        outC[j0 + u] = clast;
    }
}

// ---------------------------------------------------------------------------
// Workspace layout (bytes). Peak ~383 MB (gx is fp32, reused by both layers).
// ---------------------------------------------------------------------------
#define WS_XBF    ((size_t)0)
#define WS_WIH0   (WS_XBF   + (size_t)TT * DD * 2)
#define WS_WHH0   (WS_WIH0  + (size_t)H4 * DD * 2)
#define WS_WIH1   (WS_WHH0  + (size_t)H4 * HH * 2)
#define WS_WHH1   (WS_WIH1  + (size_t)H4 * HH * 2)
#define WS_WMU    (WS_WHH1  + (size_t)H4 * HH * 2)
#define WS_WLV    (WS_WMU   + (size_t)OO * HH * 2)
#define WS_B0     (WS_WLV   + (size_t)OO * HH * 2)
#define WS_B1     (WS_B0    + (size_t)H4 * 4)
#define WS_HBUF0  (WS_B1    + (size_t)H4 * 4)
#define WS_HBUF1  (WS_HBUF0 + (size_t)HH * 2)
#define WS_SYNC   (WS_HBUF1 + (size_t)HH * 2)
#define WS_GX     (WS_SYNC  + (size_t)256)
#define WS_H0SEQ  (WS_GX    + (size_t)TT * H4 * 4)
#define WS_H1SEQ  (WS_H0SEQ + (size_t)TT * HH * 2)

extern "C" void kernel_launch(void* const* d_in, const int* in_sizes, int n_in,
                              void* d_out, int out_size, void* d_ws, size_t ws_size,
                              hipStream_t stream) {
    const float* x    = (const float*)d_in[0];
    const float* Wih0 = (const float*)d_in[1];
    const float* Whh0 = (const float*)d_in[2];
    const float* bih0 = (const float*)d_in[3];
    const float* bhh0 = (const float*)d_in[4];
    const float* Wih1 = (const float*)d_in[5];
    const float* Whh1 = (const float*)d_in[6];
    const float* bih1 = (const float*)d_in[7];
    const float* bhh1 = (const float*)d_in[8];
    const float* Wmu  = (const float*)d_in[9];
    const float* bmu  = (const float*)d_in[10];
    const float* Wlv  = (const float*)d_in[11];
    const float* blv  = (const float*)d_in[12];

    char* ws = (char*)d_ws;
    __bf16* xbf    = (__bf16*)(ws + WS_XBF);
    __bf16* wih0b  = (__bf16*)(ws + WS_WIH0);
    __bf16* whh0b  = (__bf16*)(ws + WS_WHH0);
    __bf16* wih1b  = (__bf16*)(ws + WS_WIH1);
    __bf16* whh1b  = (__bf16*)(ws + WS_WHH1);
    __bf16* wmub   = (__bf16*)(ws + WS_WMU);
    __bf16* wlvb   = (__bf16*)(ws + WS_WLV);
    float*  b0     = (float*)(ws + WS_B0);
    float*  b1     = (float*)(ws + WS_B1);
    __bf16* hbuf0  = (__bf16*)(ws + WS_HBUF0);
    __bf16* hbuf1  = (__bf16*)(ws + WS_HBUF1);
    unsigned* sync0 = (unsigned*)(ws + WS_SYNC);
    unsigned* sync1 = sync0 + 16;
    float*  gx     = (float*)(ws + WS_GX);
    __bf16* h0seq  = (__bf16*)(ws + WS_H0SEQ);
    __bf16* h1seq  = (__bf16*)(ws + WS_H1SEQ);

    float* out      = (float*)d_out;
    float* out_mus  = out;                                // [T, O]
    float* out_sig  = out + (size_t)TT * OO;              // [T, O]
    float* out_hn   = out + (size_t)2 * TT * OO;          // [2, H]
    float* out_cn   = out_hn + 2 * HH;                    // [2, H]

    // allow 128KB dynamic LDS for the scan kernel (immediate API, capture-safe)
    (void)hipFuncSetAttribute((const void*)k_lstm_scan,
                              hipFuncAttributeMaxDynamicSharedMemorySize,
                              SCAN_LDS_BYTES);

    // --- 0) zero live-h buffers and barrier counters (re-done every call) ---
    k_zero<<<4, 256, 0, stream>>>((unsigned char*)(ws + WS_HBUF0),
                                  (long)(WS_GX - WS_HBUF0));

    // --- 1) fp32 -> bf16 conversions --------------------------------------
    k_cvt_bf16<<<2048, 256, 0, stream>>>(x,    xbf,   (long)TT * DD);
    k_cvt_bf16<<<2048, 256, 0, stream>>>(Wih0, wih0b, (long)H4 * DD);
    k_cvt_bf16<<<2048, 256, 0, stream>>>(Whh0, whh0b, (long)H4 * HH);
    k_cvt_bf16<<<2048, 256, 0, stream>>>(Wih1, wih1b, (long)H4 * HH);
    k_cvt_bf16<<<2048, 256, 0, stream>>>(Whh1, whh1b, (long)H4 * HH);
    k_cvt_bf16<<<512,  256, 0, stream>>>(Wmu,  wmub,  (long)OO * HH);
    k_cvt_bf16<<<512,  256, 0, stream>>>(Wlv,  wlvb,  (long)OO * HH);
    k_addbias<<<16, 256, 0, stream>>>(bih0, bhh0, b0, H4);
    k_addbias<<<16, 256, 0, stream>>>(bih1, bhh1, b1, H4);

    // --- 2) layer-0 input GEMM: gx = x @ Wih0^T + (bih0+bhh0) -------------
    {   const int tiles = (TT >> 4) * (H4 >> 4);           // 262144
        k_gemm_bf16<<<tiles / 8, 256, 0, stream>>>(xbf, wih0b, b0, gx,
                                                   TT, H4, DD, 0);
    }
    // --- 3) layer-0 recurrent scan (TDM-staged Whh in LDS) ----------------
    k_lstm_scan<<<HH / 16, 256, SCAN_LDS_BYTES, stream>>>(
        gx, whh0b, h0seq, hbuf0, out_hn, out_cn, sync0, TT);
    // --- 4) layer-1 input GEMM (reuses gx buffer) -------------------------
    {   const int tiles = (TT >> 4) * (H4 >> 4);
        k_gemm_bf16<<<tiles / 8, 256, 0, stream>>>(h0seq, wih1b, b1, gx,
                                                   TT, H4, HH, 0);
    }
    // --- 5) layer-1 recurrent scan ----------------------------------------
    k_lstm_scan<<<HH / 16, 256, SCAN_LDS_BYTES, stream>>>(
        gx, whh1b, h1seq, hbuf1, out_hn + HH, out_cn + HH, sync1, TT);
    // --- 6) heads: mus and sigmas -----------------------------------------
    {   const int tiles = (TT >> 4) * (OO >> 4);           // 16384
        k_gemm_bf16<<<tiles / 8, 256, 0, stream>>>(h1seq, wmub, bmu, out_mus,
                                                   TT, OO, HH, 0);
        k_gemm_bf16<<<tiles / 8, 256, 0, stream>>>(h1seq, wlvb, blv, out_sig,
                                                   TT, OO, HH, 2);
    }
}
